// SlotAttention_39986145526364
// MI455X (gfx1250) — matrix operand order repaired and checked by reading, not compiled, not run on hardware
//
#include <hip/hip_runtime.h>

// ---------------------------------------------------------------------------
// Types for CDNA5 WMMA (gfx1250, wave32)
// ---------------------------------------------------------------------------
typedef __bf16 bf16x16 __attribute__((ext_vector_type(16)));
typedef __bf16 bf16x8  __attribute__((ext_vector_type(8)));
typedef float  v8f     __attribute__((ext_vector_type(8)));

#define GEMM_ASYNC 1   // use global_load_async_to_lds_b128 staging

// Low 32 bits of a flat pointer into LDS == LDS byte address (ISA 10.2:
// LDS aperture maps addr[31:0] directly to the LDS offset).
__device__ __forceinline__ unsigned lds_off(const void* p) {
  return (unsigned)(unsigned long long)p;
}

__device__ __forceinline__ void async_b128(unsigned lds, const void* g) {
#if GEMM_ASYNC
  asm volatile("global_load_async_to_lds_b128 %0, %1, off"
               :: "v"(lds), "v"((unsigned long long)g) : "memory");
#endif
}

__device__ __forceinline__ void wait_async_4() {
#if GEMM_ASYNC
#if defined(__has_builtin) && __has_builtin(__builtin_amdgcn_s_wait_asynccnt)
  __builtin_amdgcn_s_wait_asynccnt(4);
#else
  asm volatile("s_wait_asynccnt 0x4" ::: "memory");
#endif
#endif
}
__device__ __forceinline__ void wait_async_0() {
#if GEMM_ASYNC
#if defined(__has_builtin) && __has_builtin(__builtin_amdgcn_s_wait_asynccnt)
  __builtin_amdgcn_s_wait_asynccnt(0);
#else
  asm volatile("s_wait_asynccnt 0x0" ::: "memory");
#endif
#endif
}

// ---------------------------------------------------------------------------
// conv1: 7x7x3->8, SAME pad 3, relu.  One thread per pixel, weights in LDS.
// in rgb (128,128,128,3) NHWC fp32 -> out (128,128,128,8) fp32
// ---------------------------------------------------------------------------
__global__ __launch_bounds__(256)
void conv1_kernel(const float* __restrict__ rgb, const float* __restrict__ w,
                  const float* __restrict__ b, float* __restrict__ out)
{
  __shared__ float sw[7 * 7 * 3 * 8];
  __shared__ float sb[8];
  for (int i = threadIdx.x; i < 7 * 7 * 3 * 8; i += 256) sw[i] = w[i];
  if (threadIdx.x < 8) sb[threadIdx.x] = b[threadIdx.x];
  __syncthreads();
  int gid = blockIdx.x * 256 + threadIdx.x;   // 128*128*128 pixels
  int x = gid & 127, y = (gid >> 7) & 127, img = gid >> 14;
  float acc[8];
#pragma unroll
  for (int o = 0; o < 8; ++o) acc[o] = sb[o];
  const float* src = rgb + (size_t)img * 128 * 128 * 3;
  __builtin_prefetch(src + (size_t)(y * 128 + x) * 3 + 128 * 3, 0, 1);
  for (int r = 0; r < 7; ++r) {
    int yy = y + r - 3; if (yy < 0 || yy > 127) continue;
    for (int s = 0; s < 7; ++s) {
      int xx = x + s - 3; if (xx < 0 || xx > 127) continue;
      const float* p  = src + (size_t)(yy * 128 + xx) * 3;
      const float* wp = &sw[(r * 7 + s) * 3 * 8];
#pragma unroll
      for (int c = 0; c < 3; ++c) {
        float v = p[c];
#pragma unroll
        for (int o = 0; o < 8; ++o) acc[o] = fmaf(v, wp[c * 8 + o], acc[o]);
      }
    }
  }
  float* op = out + (size_t)gid * 8;
#pragma unroll
  for (int o = 0; o < 8; ++o) op[o] = fmaxf(acc[o], 0.f);
}

// ---------------------------------------------------------------------------
// conv2: 5x5x8->8, relu, fused with the (t1 i)(t2 j)k -> t1 t2 (i j k)
// rearrange + fp32->bf16 pack of the GEMM A matrix (8192 x 2048).
// ---------------------------------------------------------------------------
__global__ __launch_bounds__(256)
void conv2_pack_kernel(const float* __restrict__ in, const float* __restrict__ w,
                       const float* __restrict__ b, __bf16* __restrict__ A)
{
  __shared__ float sw[5 * 5 * 8 * 8];
  __shared__ float sb[8];
  for (int i = threadIdx.x; i < 5 * 5 * 8 * 8; i += 256) sw[i] = w[i];
  if (threadIdx.x < 8) sb[threadIdx.x] = b[threadIdx.x];
  __syncthreads();
  int gid = blockIdx.x * 256 + threadIdx.x;
  int x = gid & 127, y = (gid >> 7) & 127, img = gid >> 14;
  float acc[8];
#pragma unroll
  for (int o = 0; o < 8; ++o) acc[o] = sb[o];
  const float* src = in + (size_t)img * 128 * 128 * 8;
  for (int r = 0; r < 5; ++r) {
    int yy = y + r - 2; if (yy < 0 || yy > 127) continue;
    for (int s = 0; s < 5; ++s) {
      int xx = x + s - 2; if (xx < 0 || xx > 127) continue;
      const float* p  = src + (size_t)(yy * 128 + xx) * 8;
      const float* wp = &sw[(r * 5 + s) * 64];
#pragma unroll
      for (int c = 0; c < 8; ++c) {
        float v = p[c];
#pragma unroll
        for (int o = 0; o < 8; ++o) acc[o] = fmaf(v, wp[c * 8 + o], acc[o]);
      }
    }
  }
  // token row m, feature col
  int t1 = y >> 4, t2 = x >> 4, ii = y & 15, jj = x & 15;
  int m   = img * 64 + t1 * 8 + t2;
  int col = (ii * 16 + jj) * 8;
  __bf16* ap = A + (size_t)m * 2048 + col;
#pragma unroll
  for (int o = 0; o < 8; ++o) ap[o] = (__bf16)fmaxf(acc[o], 0.f);
}

// ---------------------------------------------------------------------------
// W (K x N fp32, [in][out]) -> Wt (N x K bf16) so GEMM B tiles are straight
// row-major copies (async-to-LDS friendly, no transpose in the hot loop).
// ---------------------------------------------------------------------------
__global__ __launch_bounds__(256)
void packw_kernel(const float* __restrict__ W, __bf16* __restrict__ Wt, int K, int N)
{
  int gid = blockIdx.x * 256 + threadIdx.x;
  if (gid >= K * N) return;
  int n = gid % N, k = gid / N;
  Wt[(size_t)n * K + k] = (__bf16)W[(size_t)k * N + n];
}

// pos_emb[t][n] for the 8x8 patch grid: grid @ pos_w + pos_b   (64 x 2048)
__global__ __launch_bounds__(256)
void pos_kernel(const float* __restrict__ pw, const float* __restrict__ pb,
                float* __restrict__ pos)
{
  int gid = blockIdx.x * 256 + threadIdx.x;   // 64*2048
  int n = gid & 2047, t = gid >> 11;
  float g0 = (float)(t >> 3) * (1.f / 7.f);
  float g1 = (float)(t & 7)  * (1.f / 7.f);
  float v = g0 * pw[0 * 2048 + n] + g1 * pw[1 * 2048 + n]
          + (1.f - g0) * pw[2 * 2048 + n] + (1.f - g1) * pw[3 * 2048 + n] + pb[n];
  pos[(size_t)t * 2048 + n] = v;
}

// ---------------------------------------------------------------------------
// bf16 WMMA GEMM:  C[M,N] = epi( A[M,K] @ Bt[N,K]^T + bias )  (+ pos)
// 256 threads = 8 wave32; block tile 128x128; wave tile 32x64 = 2x4 WMMA.
// LDS double-buffered 2x(8KB A + 8KB B); async global->LDS staging.
// ---------------------------------------------------------------------------
#define TM 128
#define TN 128
#define TK 32

__global__ __launch_bounds__(256)
void gemm_bf16_kernel(const __bf16* __restrict__ A, const __bf16* __restrict__ Bt,
                      float* __restrict__ C, const float* __restrict__ bias,
                      const float* __restrict__ pos, int M, int N, int K, int relu)
{
  __shared__ __align__(16) __bf16 sA[2][TM][TK];
  __shared__ __align__(16) __bf16 sB[2][TN][TK];

  const int tid  = threadIdx.x;
  const int wave = tid >> 5;
  const int lane = tid & 31;
  const int bm   = blockIdx.x;      // M / 128
  const int bn   = blockIdx.y;      // N / 128
  const int wm   = wave & 3;        // 32-row slab
  const int wn   = wave >> 2;       // 64-col slab

  auto load_tiles = [&](int buf, int kt) {
#pragma unroll
    for (int itc = 0; itc < 2; ++itc) {
      int c   = tid + itc * 256;       // 512 16B chunks per tile
      int row = c >> 2;
      int ko  = (c & 3) * 8;
      const __bf16* gA = A  + (size_t)(bm * TM + row) * K + (size_t)kt * TK + ko;
      const __bf16* gB = Bt + (size_t)(bn * TN + row) * K + (size_t)kt * TK + ko;
#if GEMM_ASYNC
      async_b128(lds_off(&sA[buf][row][ko]), gA);
      async_b128(lds_off(&sB[buf][row][ko]), gB);
#else
      *(bf16x8*)&sA[buf][row][ko] = *(const bf16x8*)gA;
      *(bf16x8*)&sB[buf][row][ko] = *(const bf16x8*)gB;
#endif
    }
  };

  v8f acc[2][4];
#pragma unroll
  for (int i = 0; i < 2; ++i)
#pragma unroll
    for (int j = 0; j < 4; ++j)
#pragma unroll
      for (int r = 0; r < 8; ++r) acc[i][j][r] = 0.f;

  const int KT = K / TK;
  load_tiles(0, 0);

  for (int kt = 0; kt < KT; ++kt) {
    const int cur = kt & 1;
    if (kt + 1 < KT) { load_tiles(cur ^ 1, kt + 1); wait_async_4(); }
    else             { wait_async_0(); }
    __syncthreads();

    // A fragments: lane L in [0,16): M=L, elems[0..7]=K(kb..kb+7),
    // elems[8..15]=K(kb+16..kb+23), kb = (lane>=16)*8  (ISA 7.12.2)
    bf16x16 af[2];
    const int ka = (lane >> 4) * 8;
#pragma unroll
    for (int i = 0; i < 2; ++i) {
      const __bf16* ar = &sA[cur][wm * 32 + i * 16 + (lane & 15)][0];
      bf16x8 lo = *(const bf16x8*)(ar + ka);
      bf16x8 hi = *(const bf16x8*)(ar + ka + 16);
      af[i] = __builtin_shufflevector(lo, hi, 0,1,2,3,4,5,6,7,8,9,10,11,12,13,14,15);
    }
    // B fragments: lane = N column, same K layout as A (kb..kb+7, kb+16..kb+23)
    bf16x16 bfr[4];
    const int kb = (lane >> 4) * 8;
#pragma unroll
    for (int j = 0; j < 4; ++j) {
      const __bf16* br = &sB[cur][wn * 64 + j * 16 + (lane & 15)][0];
      bf16x8 lo = *(const bf16x8*)(br + kb);
      bf16x8 hi = *(const bf16x8*)(br + kb + 16);
      bfr[j] = __builtin_shufflevector(lo, hi, 0,1,2,3,4,5,6,7,8,9,10,11,12,13,14,15);
    }

#pragma unroll
    for (int i = 0; i < 2; ++i)
#pragma unroll
      for (int j = 0; j < 4; ++j)
        acc[i][j] = __builtin_amdgcn_wmma_f32_16x16x32_bf16(
            false, af[i], false, bfr[j], (short)0, acc[i][j], false, false);

    __syncthreads();
  }

  // Epilogue: C/D layout -> row = base + r + 8*(lane>=16), col = lane&15
#pragma unroll
  for (int i = 0; i < 2; ++i) {
#pragma unroll
    for (int j = 0; j < 4; ++j) {
      int row0 = bm * TM + wm * 32 + i * 16 + ((lane < 16) ? 0 : 8);
      int col  = bn * TN + wn * 64 + j * 16 + (lane & 15);
      float bv = bias[col];
#pragma unroll
      for (int r = 0; r < 8; ++r) {
        int row = row0 + r;
        float x = acc[i][j][r] + bv;
        if (relu) x = fmaxf(x, 0.f);
        if (pos)  x += pos[(size_t)(row & 63) * N + col];
        C[(size_t)row * N + col] = x;
      }
    }
  }
}

// ---------------------------------------------------------------------------
// LayerNorm over 2048 per token row, output bf16 (feeds mlp1 WMMA GEMM).
// ---------------------------------------------------------------------------
__global__ __launch_bounds__(256)
void ln2048_kernel(const float* __restrict__ X, const float* __restrict__ gs,
                   const float* __restrict__ gb, __bf16* __restrict__ Y)
{
  int row = blockIdx.x, tid = threadIdx.x;
  const float* x = X + (size_t)row * 2048;
  float sum = 0.f, ss = 0.f;
  for (int i = tid; i < 2048; i += 256) { float v = x[i]; sum += v; ss += v * v; }
  __shared__ float r1[256], r2[256];
  r1[tid] = sum; r2[tid] = ss; __syncthreads();
  for (int off = 128; off; off >>= 1) {
    if (tid < off) { r1[tid] += r1[tid + off]; r2[tid] += r2[tid + off]; }
    __syncthreads();
  }
  float mu  = r1[0] * (1.f / 2048.f);
  float var = r2[0] * (1.f / 2048.f) - mu * mu;
  float inv = rsqrtf(var + 1e-6f);
  __bf16* y = Y + (size_t)row * 2048;
  for (int i = tid; i < 2048; i += 256)
    y[i] = (__bf16)((x[i] - mu) * inv * gs[i] + gb[i]);
}

// ---------------------------------------------------------------------------
// Per-row (8192 rows, 128 threads): h2 = h1@mlp2+b, LN1, k = xln@k_w, v = xln@v_w
// ---------------------------------------------------------------------------
__global__ __launch_bounds__(128)
void post_kernel(const float* __restrict__ H1, const float* __restrict__ w2,
                 const float* __restrict__ b2, const float* __restrict__ lns,
                 const float* __restrict__ lnb, const float* __restrict__ kw,
                 const float* __restrict__ vw, float* __restrict__ Kout,
                 float* __restrict__ Vout)
{
  int row = blockIdx.x, tid = threadIdx.x;
  __shared__ float h1[128], xln[128], r1[128], r2[128];
  h1[tid] = H1[(size_t)row * 128 + tid];
  __syncthreads();
  float a = b2[tid];
  for (int i = 0; i < 128; ++i) a = fmaf(h1[i], w2[i * 128 + tid], a);
  r1[tid] = a; r2[tid] = a * a; __syncthreads();
  for (int off = 64; off; off >>= 1) {
    if (tid < off) { r1[tid] += r1[tid + off]; r2[tid] += r2[tid + off]; }
    __syncthreads();
  }
  float mu  = r1[0] * (1.f / 128.f);
  float var = r2[0] * (1.f / 128.f) - mu * mu;
  float inv = rsqrtf(var + 1e-6f);
  xln[tid] = (a - mu) * inv * lns[tid] + lnb[tid];
  __syncthreads();
  int c = tid & 63;
  const float* w = (tid < 64) ? kw : vw;
  float acc = 0.f;
  for (int i = 0; i < 128; ++i) acc = fmaf(xln[i], w[i * 64 + c], acc);
  float* out = (tid < 64) ? Kout : Vout;
  out[(size_t)row * 64 + c] = acc;
}

// LN over 4 slots x 64 dims, all 256 threads participate (s=tid>>6, d=tid&63)
__device__ __forceinline__ void ln4x64(float (*src)[64], float (*dst)[64],
                                       const float* gs, const float* gb,
                                       float (*red)[64], int s, int d)
{
  red[s][d] = src[s][d];
  __syncthreads();
  for (int off = 32; off; off >>= 1) { if (d < off) red[s][d] += red[s][d + off]; __syncthreads(); }
  float mu = red[s][0] * (1.f / 64.f);
  __syncthreads();
  float c = src[s][d] - mu;
  red[s][d] = c * c;
  __syncthreads();
  for (int off = 32; off; off >>= 1) { if (d < off) red[s][d] += red[s][d + off]; __syncthreads(); }
  float inv = rsqrtf(red[s][0] * (1.f / 64.f) + 1e-6f);
  __syncthreads();
  dst[s][d] = c * inv * gs[d] + gb[d];
  __syncthreads();
}

// ---------------------------------------------------------------------------
// Slot attention: one workgroup per image (128 blocks, 256 threads), 3 iters.
// k,v (64x64) resident in LDS; slots/attn/GRU all in LDS.
// ---------------------------------------------------------------------------
__global__ __launch_bounds__(256)
void slot_kernel(const float* __restrict__ Kin, const float* __restrict__ Vin,
                 const float* __restrict__ noise,
                 const float* __restrict__ init_mu, const float* __restrict__ init_ls,
                 const float* __restrict__ ln2s, const float* __restrict__ ln2b,
                 const float* __restrict__ qw,
                 const float* __restrict__ gwi, const float* __restrict__ gbi,
                 const float* __restrict__ gwh, const float* __restrict__ gbhn,
                 const float* __restrict__ ln3s, const float* __restrict__ ln3b,
                 const float* __restrict__ s1w, const float* __restrict__ s1b,
                 const float* __restrict__ s2w, const float* __restrict__ s2b,
                 float* __restrict__ slots_out)
{
  int b = blockIdx.x, tid = threadIdx.x;
  __shared__ float k[64][64], v[64][64];
  __shared__ float slots[4][64], prev[4][64], sln[4][64], q[4][64];
  __shared__ float attn[64][4], colsum[4];
  __shared__ float upd[4][64], red[4][64];
  __shared__ float gi[4][192], gh[4][192];
  __shared__ float hbuf[4][128];

  for (int i = tid; i < 4096; i += 256) {
    k[i >> 6][i & 63] = Kin[(size_t)b * 4096 + i];
    v[i >> 6][i & 63] = Vin[(size_t)b * 4096 + i];
  }
  int s = tid >> 6, d = tid & 63;
  slots[s][d] = init_mu[d] + noise[(size_t)b * 256 + tid] * expf(init_ls[d]);
  __syncthreads();

  const float scale = 0.08838834764831845f;   // 128^-0.5
  for (int it = 0; it < 3; ++it) {
    prev[s][d] = slots[s][d];
    __syncthreads();
    ln4x64(slots, sln, ln2s, ln2b, red, s, d);
    { float a = 0.f;
      for (int i = 0; i < 64; ++i) a = fmaf(sln[s][i], qw[i * 64 + d], a);
      q[s][d] = a; }
    __syncthreads();
    { int n = tid >> 2, ss = tid & 3;
      float a = 0.f;
      for (int i = 0; i < 64; ++i) a = fmaf(k[n][i], q[ss][i], a);
      attn[n][ss] = a * scale; }
    __syncthreads();
    if (tid < 64) {
      float m = attn[tid][0];
      for (int j = 1; j < 4; ++j) m = fmaxf(m, attn[tid][j]);
      float e[4], sum = 0.f;
      for (int j = 0; j < 4; ++j) { e[j] = expf(attn[tid][j] - m); sum += e[j]; }
      for (int j = 0; j < 4; ++j) attn[tid][j] = e[j] / sum + 1e-7f;
    }
    __syncthreads();
    if (tid < 4) { float t = 0.f; for (int n = 0; n < 64; ++n) t += attn[n][tid]; colsum[tid] = t; }
    __syncthreads();
    { int n = tid >> 2, ss = tid & 3; attn[n][ss] /= colsum[ss]; }
    __syncthreads();
    { float a = 0.f;
      for (int n = 0; n < 64; ++n) a = fmaf(attn[n][s], v[n][d], a);
      upd[s][d] = a; }
    __syncthreads();
    for (int rep = 0; rep < 3; ++rep) {            // 768 GRU pre-activations
      int idx = tid + rep * 256, s2 = idx / 192, j = idx - s2 * 192;
      float a = gbi[j], h = 0.f;
      for (int i = 0; i < 64; ++i) {
        a = fmaf(upd[s2][i],  gwi[i * 192 + j], a);
        h = fmaf(prev[s2][i], gwh[i * 192 + j], h);
      }
      gi[s2][j] = a; gh[s2][j] = h;
    }
    __syncthreads();
    { float r  = 1.f / (1.f + expf(-(gi[s][d]      + gh[s][d])));
      float z  = 1.f / (1.f + expf(-(gi[s][64 + d] + gh[s][64 + d])));
      float nn = tanhf(gi[s][128 + d] + r * (gh[s][128 + d] + gbhn[d]));
      slots[s][d] = (1.f - z) * nn + z * prev[s][d]; }
    __syncthreads();
    ln4x64(slots, sln, ln3s, ln3b, red, s, d);
    for (int rep = 0; rep < 2; ++rep) {            // 512 hidden
      int idx = tid + rep * 256, s2 = idx >> 7, j = idx & 127;
      float a = s1b[j];
      for (int i = 0; i < 64; ++i) a = fmaf(sln[s2][i], s1w[i * 128 + j], a);
      hbuf[s2][j] = fmaxf(a, 0.f);
    }
    __syncthreads();
    { float a = s2b[d];
      for (int j = 0; j < 128; ++j) a = fmaf(hbuf[s][j], s2w[j * 64 + d], a);
      slots[s][d] += a; }
    __syncthreads();
  }
  slots_out[(size_t)b * 256 + tid] = slots[s][d];
}

// ---------------------------------------------------------------------------
// Heads: one workgroup per (image, slot); 64->64->64->3, tanh + affine.
// ---------------------------------------------------------------------------
__global__ __launch_bounds__(64)
void head_kernel(const float* __restrict__ slots,
                 const float* __restrict__ w1, const float* __restrict__ b1,
                 const float* __restrict__ w2, const float* __restrict__ b2,
                 const float* __restrict__ w3, const float* __restrict__ b3,
                 float* __restrict__ out)
{
  int bs = blockIdx.x, t = threadIdx.x;
  __shared__ float s0[64], h1[64], h2[64];
  s0[t] = slots[(size_t)bs * 64 + t];
  __syncthreads();
  float a = b1[t];
  for (int i = 0; i < 64; ++i) a = fmaf(s0[i], w1[i * 64 + t], a);
  h1[t] = fmaxf(a, 0.f);
  __syncthreads();
  a = b2[t];
  for (int i = 0; i < 64; ++i) a = fmaf(h1[i], w2[i * 64 + t], a);
  h2[t] = fmaxf(a, 0.f);
  __syncthreads();
  if (t < 3) {
    float o = b3[t];
    for (int i = 0; i < 64; ++i) o = fmaf(h2[i], w3[i * 3 + t], o);
    o = tanhf(o) + ((t == 0) ? 1.f : 0.f);
    o *= ((t == 0) ? 0.5f : 0.2f);
    out[(size_t)bs * 3 + t] = o;
  }
}

// ---------------------------------------------------------------------------
// Host launcher
// ---------------------------------------------------------------------------
extern "C" void kernel_launch(void* const* d_in, const int* in_sizes, int n_in,
                              void* d_out, int out_size, void* d_ws, size_t ws_size,
                              hipStream_t stream)
{
  (void)in_sizes; (void)n_in; (void)out_size; (void)ws_size;
  const float* rgb      = (const float*)d_in[0];
  const float* noise    = (const float*)d_in[1];
  const float* conv1_w  = (const float*)d_in[2];
  const float* conv1_b  = (const float*)d_in[3];
  const float* conv2_w  = (const float*)d_in[4];
  const float* conv2_b  = (const float*)d_in[5];
  const float* conv3_w  = (const float*)d_in[6];
  const float* conv3_b  = (const float*)d_in[7];
  const float* pos_w    = (const float*)d_in[8];
  const float* pos_b    = (const float*)d_in[9];
  const float* in_ln_s  = (const float*)d_in[10];
  const float* in_ln_b  = (const float*)d_in[11];
  const float* mlp1_w   = (const float*)d_in[12];
  const float* mlp1_b   = (const float*)d_in[13];
  const float* mlp2_w   = (const float*)d_in[14];
  const float* mlp2_b   = (const float*)d_in[15];
  const float* ln1_s    = (const float*)d_in[16];
  const float* ln1_b    = (const float*)d_in[17];
  const float* k_w      = (const float*)d_in[18];
  const float* v_w      = (const float*)d_in[19];
  const float* q_w      = (const float*)d_in[20];
  const float* ln2_s    = (const float*)d_in[21];
  const float* ln2_b    = (const float*)d_in[22];
  const float* ln3_s    = (const float*)d_in[23];
  const float* ln3_b    = (const float*)d_in[24];
  const float* gru_wi   = (const float*)d_in[25];
  const float* gru_bi   = (const float*)d_in[26];
  const float* gru_wh   = (const float*)d_in[27];
  const float* gru_bhn  = (const float*)d_in[28];
  const float* smlp1_w  = (const float*)d_in[29];
  const float* smlp1_b  = (const float*)d_in[30];
  const float* smlp2_w  = (const float*)d_in[31];
  const float* smlp2_b  = (const float*)d_in[32];
  const float* init_mu  = (const float*)d_in[33];
  const float* init_ls  = (const float*)d_in[34];
  const float* h1w      = (const float*)d_in[35];
  const float* h1b      = (const float*)d_in[36];
  const float* h2w      = (const float*)d_in[37];
  const float* h2b      = (const float*)d_in[38];
  const float* h3w      = (const float*)d_in[39];
  const float* h3b      = (const float*)d_in[40];

  char* ws = (char*)d_ws;
  const size_t MB = 1ull << 20;
  float*  buf1  = (float*)(ws);                 // 64 MB: conv1 out, reused as tokens
  __bf16* Abf   = (__bf16*)(ws + 64 * MB);      // 32 MB: GEMM A, reused as xln bf16
  __bf16* Wt    = (__bf16*)(ws + 96 * MB);      //  8 MB: packed weights (bf16, NxK)
  float*  pos   = (float*)(ws + 104 * MB);      // 0.5 MB
  float*  h1    = (float*)(ws + 105 * MB);      //  4 MB
  float*  kbuf  = (float*)(ws + 109 * MB);      //  2 MB
  float*  vbuf  = (float*)(ws + 111 * MB);      //  2 MB
  float*  slots = (float*)(ws + 113 * MB);      // 128 KB
  float*  tokens = buf1;                        // conv1-out dead after conv2
  __bf16* xln    = Abf;                         // A dead after conv3 GEMM

  conv1_kernel<<<8192, 256, 0, stream>>>(rgb, conv1_w, conv1_b, buf1);
  conv2_pack_kernel<<<8192, 256, 0, stream>>>(buf1, conv2_w, conv2_b, Abf);
  packw_kernel<<<(2048 * 2048) / 256, 256, 0, stream>>>(conv3_w, Wt, 2048, 2048);
  pos_kernel<<<(64 * 2048) / 256, 256, 0, stream>>>(pos_w, pos_b, pos);
  // conv3 as WMMA GEMM: (8192x2048)@(2048x2048), relu + positional epilogue
  gemm_bf16_kernel<<<dim3(64, 16), 256, 0, stream>>>(Abf, Wt, tokens, conv3_b, pos,
                                                     8192, 2048, 2048, 1);
  ln2048_kernel<<<8192, 256, 0, stream>>>(tokens, in_ln_s, in_ln_b, xln);
  packw_kernel<<<(2048 * 128) / 256, 256, 0, stream>>>(mlp1_w, Wt, 2048, 128);
  // mlp1 as WMMA GEMM: (8192x2048)@(2048x128), relu epilogue
  gemm_bf16_kernel<<<dim3(64, 1), 256, 0, stream>>>(xln, Wt, h1, mlp1_b, nullptr,
                                                    8192, 128, 2048, 1);
  post_kernel<<<8192, 128, 0, stream>>>(h1, mlp2_w, mlp2_b, ln1_s, ln1_b,
                                        k_w, v_w, kbuf, vbuf);
  slot_kernel<<<128, 256, 0, stream>>>(kbuf, vbuf, noise, init_mu, init_ls,
                                       ln2_s, ln2_b, q_w, gru_wi, gru_bi, gru_wh,
                                       gru_bhn, ln3_s, ln3_b, smlp1_w, smlp1_b,
                                       smlp2_w, smlp2_b, slots);
  head_kernel<<<512, 64, 0, stream>>>(slots, h1w, h1b, h2w, h2b, h3w, h3b,
                                      (float*)d_out);
}
